// SimpleGNN_1872605741404
// MI455X (gfx1250) — compile-verified, matrix operand order checked by reading, not run on hardware
//
#include <hip/hip_runtime.h>
#include <hip/hip_bf16.h>
#include <math.h>

typedef __attribute__((ext_vector_type(16))) _Float16 v16h;
typedef __attribute__((ext_vector_type(8)))  _Float16 v8h;
typedef __attribute__((ext_vector_type(8)))  float    v8f;

#define K_DIM 128   // inner dim for both GEMMs (IN_DIM == HID_DIM == 128)

// ---- gfx1250 async global->LDS copy (ASYNCcnt-tracked), per-lane 16B ----
__device__ __forceinline__ void async_g2l_b128(unsigned lds_off, const void* gaddr) {
  asm volatile("global_load_async_to_lds_b128 %0, %1, off"
               :: "v"(lds_off), "v"(gaddr)
               : "memory");
}

__device__ __forceinline__ void wait_asynccnt0() {
#if __has_builtin(__builtin_amdgcn_s_wait_asynccnt)
  __builtin_amdgcn_s_wait_asynccnt(0);
#else
  asm volatile("s_wait_asynccnt 0x0" ::: "memory");
#endif
}

// ---------------- elementwise converts ----------------
__global__ void k_f32_to_f16(const float* __restrict__ in, _Float16* __restrict__ out, long n) {
  long i = (long)blockIdx.x * blockDim.x + threadIdx.x;
  if (i < n) out[i] = (_Float16)in[i];
}

__global__ void k_relu_f32_to_f16(const float* __restrict__ in, _Float16* __restrict__ out, long n) {
  long i = (long)blockIdx.x * blockDim.x + threadIdx.x;
  if (i < n) {
    float v = in[i];
    out[i] = (_Float16)(v > 0.0f ? v : 0.0f);
  }
}

// ---- one-shot: convert f32 weights [128 x F] into f16 packed WMMA-B layout ----
// Packed entry e = (colTile*4 + kblock)*32 + lane holds 16 contiguous halves:
// j<8 -> K = kb*32 + hi*8 + j ; j>=8 -> +16.  (ISA 7.12.2, 16-bit B striping)
__global__ void k_pack_Bf(const float* __restrict__ B, _Float16* __restrict__ P, int F) {
  int idx = blockIdx.x * blockDim.x + threadIdx.x;
  int total = F * 128;
  if (idx >= total) return;
  int e  = idx >> 4;
  int j  = idx & 15;
  int l  = e & 31;
  int kb = (e >> 5) & 3;
  int c  = e >> 7;
  int m  = l & 15;
  int hi = l >> 4;
  int col = c * 16 + m;
  int k = kb * 32 + hi * 8 + (j & 7) + ((j >> 3) ? 16 : 0);
  P[idx] = (_Float16)B[k * F + col];
}

// ---------------- degree / normalization ----------------
__global__ void k_init_deg(float* __restrict__ deg, int n) {
  int i = blockIdx.x * blockDim.x + threadIdx.x;
  if (i < n) deg[i] = 1.0f;   // self-loop contributes 1
}

__global__ void k_accum_deg(const int* __restrict__ dst, float* __restrict__ deg, int E) {
  int e = blockIdx.x * blockDim.x + threadIdx.x;
  if (e < E)
    (void)__hip_atomic_fetch_add(&deg[dst[e]], 1.0f, __ATOMIC_RELAXED, __HIP_MEMORY_SCOPE_AGENT);
}

__global__ void k_finalize_dinv(float* __restrict__ deg, int n) {
  int i = blockIdx.x * blockDim.x + threadIdx.x;
  if (i < n) deg[i] = rsqrtf(deg[i]);  // deg >= 1 always
}

// ---------------- WMMA GEMM: C[N x F] = A[N x 128] * Bpacked ----------------
// One wave computes one 16x16 output tile; K traversed in 4 steps of 32 with
// v_wmma_f32_16x16x32_f16. Pre-packed B image streamed global->LDS with
// async-to-LDS b128 ops (ASYNCcnt), then read back as ds_load_b128 pairs.
__global__ void k_gemm_wmma(const _Float16* __restrict__ A,
                            const _Float16* __restrict__ Bp,  // packed, F*128 halves
                            float* __restrict__ C,
                            int nrows, int F) {
  __shared__ _Float16 smB[16384];  // up to 32KB

  // Stage packed B: contiguous copy, 16B per async op per lane. bytes is a
  // multiple of blockDim.x*16 (F=128 -> 32768, F=64 -> 16384), so EXEC is
  // all-ones for every issued async op.
  int bytes = F * 256;
  unsigned ldsBase = (unsigned)(unsigned long long)(const void*)smB;  // LDS offset = low 32 bits
  for (int ofs = threadIdx.x * 16; ofs < bytes; ofs += blockDim.x * 16) {
    async_g2l_b128(ldsBase + (unsigned)ofs, (const char*)Bp + ofs);
  }
  wait_asynccnt0();
  __syncthreads();

  int wave = (blockIdx.x * blockDim.x + threadIdx.x) >> 5;
  int lane = threadIdx.x & 31;
  int tiles = F >> 4;
  int rowTile = wave / tiles;
  int colTile = wave % tiles;
  if (rowTile * 16 >= nrows) return;  // wave-uniform

  int m  = lane & 15;
  int hi = lane >> 4;
  const _Float16* Arow = A + (long)(rowTile * 16 + m) * K_DIM;

  v8f acc = {};
#pragma unroll
  for (int kb = 0; kb < 4; ++kb) {
    int k0 = kb * 32 + hi * 8;
    // A: lane m holds row (tile,m); K runs {k0..k0+7} and {k0+16..k0+23}, 16B each
    v8h a0 = *(const v8h*)(Arow + k0);
    v8h a1 = *(const v8h*)(Arow + k0 + 16);
    v16h av = __builtin_shufflevector(a0, a1, 0,1,2,3,4,5,6,7,8,9,10,11,12,13,14,15);
    // B: pre-packed in LDS, two contiguous 16B chunks per lane
    const _Float16* bp = smB + ((long)((colTile * 4 + kb) * 32 + lane)) * 16;
    v8h b0 = *(const v8h*)bp;
    v8h b1 = *(const v8h*)(bp + 8);
    v16h bv = __builtin_shufflevector(b0, b1, 0,1,2,3,4,5,6,7,8,9,10,11,12,13,14,15);
    acc = __builtin_amdgcn_wmma_f32_16x16x32_f16(false, av, false, bv,
                                                 (short)0, acc, false, false);
  }

  // D layout: VGPR i -> M = hi*8 + i, N = m
  int crow = rowTile * 16 + hi * 8;
  int ccol = colTile * 16 + m;
  float* Cp = C + (long)crow * F + ccol;
#pragma unroll
  for (int i = 0; i < 8; ++i) Cp[(long)i * F] = acc[i];
}

// ---------------- aggregation init: agg = bias + h * dinv^2 (self loop) ----------------
__global__ void k_init_agg(const float* __restrict__ h, const float* __restrict__ dinv,
                           const float* __restrict__ bias, float* __restrict__ agg,
                           long n, int F) {
  long idx = (long)blockIdx.x * blockDim.x + threadIdx.x;
  if (idx >= n * (long)F) return;
  long i = idx / F;
  int  f = (int)(idx - i * F);
  float di = dinv[i];
  agg[idx] = bias[f] + h[idx] * di * di;
}

// ---------------- edge scatter: one wave per edge, lane-striped features ----------------
__global__ void k_edge_scatter(const int* __restrict__ src, const int* __restrict__ dst,
                               const float* __restrict__ dinv,
                               const float* __restrict__ h, float* __restrict__ agg,
                               int E, int F) {
  int wid  = (blockIdx.x * blockDim.x + threadIdx.x) >> 5;
  int lane = threadIdx.x & 31;
  if (wid >= E) return;
  int s = src[wid];
  int d = dst[wid];
  float norm = dinv[s] * dinv[d];
  const float* hs = h + (long)s * F;
  float*       ad = agg + (long)d * F;
  int per = F >> 5;  // 4 (F=128) or 2 (F=64)
  for (int j = 0; j < per; ++j) {
    int f = lane + j * 32;  // coalesced
    (void)__hip_atomic_fetch_add(&ad[f], hs[f] * norm,
                                 __ATOMIC_RELAXED, __HIP_MEMORY_SCOPE_AGENT);
  }
}

// ---------------- sigmoid + deterministic mean over nodes ----------------
__global__ void k_sigmoid_partial(const float* __restrict__ agg, float* __restrict__ part, int n) {
  __shared__ float sm[256];
  int f   = threadIdx.x & 63;
  int sub = threadIdx.x >> 6;  // 0..3
  int base = blockIdx.x * 256;
  int end  = base + 256; if (end > n) end = n;
  float s = 0.0f;
  for (int i = base + sub; i < end; i += 4) {
    float v = agg[(long)i * 64 + f];
    s += 1.0f / (1.0f + expf(-v));
  }
  sm[threadIdx.x] = s;
  __syncthreads();
  if (sub == 0)
    part[(long)blockIdx.x * 64 + f] = sm[f] + sm[64 + f] + sm[128 + f] + sm[192 + f];
}

__global__ void k_mean_final(const float* __restrict__ part, float* __restrict__ out,
                             int nb, float invN) {
  int f = threadIdx.x;  // 64 threads
  float s = 0.0f;
  for (int b = 0; b < nb; ++b) s += part[(long)b * 64 + f];
  out[f] = s * invN;
}

// ---------------- host side ----------------
extern "C" void kernel_launch(void* const* d_in, const int* in_sizes, int n_in,
                              void* d_out, int out_size, void* d_ws, size_t ws_size,
                              hipStream_t stream) {
  (void)n_in; (void)out_size; (void)ws_size;

  const float* x  = (const float*)d_in[0];   // [N,128]
  const int*   ei = (const int*)d_in[1];     // [2,E]
  const float* W1 = (const float*)d_in[2];   // [128,128]
  const float* b1 = (const float*)d_in[3];   // [128]
  const float* W2 = (const float*)d_in[4];   // [128,64]
  const float* b2 = (const float*)d_in[5];   // [64]
  float* out = (float*)d_out;                // [64]

  const long N = in_sizes[0] / 128;
  const long E = in_sizes[1] / 2;
  const int* src = ei;
  const int* dst = ei + E;

  // Workspace carving (256B aligned), with buffer reuse.
  char* w = (char*)d_ws;
  size_t off = 0;
  auto carve = [&](size_t bytes) -> void* {
    void* p = w + off;
    off = (off + bytes + 255) & ~(size_t)255;
    return p;
  };
  float*    dinv = (float*)carve((size_t)N * 4);
  _Float16* P1   = (_Float16*)carve(128 * 128 * 2);  // packed W1 (f16)
  _Float16* P2   = (_Float16*)carve(128 * 64 * 2);   // packed W2 (f16)
  int nbMean = (int)((N + 255) / 256);
  float*    part = (float*)carve((size_t)nbMean * 64 * 4);
  _Float16* Xh   = (_Float16*)carve((size_t)N * 128 * 2);
  float*    H1   = (float*)carve((size_t)N * 128 * 4);
  float*    AGG1 = (float*)carve((size_t)N * 128 * 4);
  _Float16* RELU = Xh;    // Xh dead after GEMM1
  float*    H2   = H1;    // H1 dead after layer-1 scatter
  float*    AGG2 = AGG1;  // AGG1 dead after relu-convert

  const int T = 256;
  auto g = [](long n, int t) -> unsigned { return (unsigned)((n + t - 1) / t); };

  // f16 copy of X; weights converted+packed into WMMA-B layout in one shot
  k_f32_to_f16<<<g(N * 128, T), T, 0, stream>>>(x, Xh, N * 128);
  k_pack_Bf<<<g(128 * 128, T), T, 0, stream>>>(W1, P1, 128);
  k_pack_Bf<<<g(128 * 64,  T), T, 0, stream>>>(W2, P2, 64);

  // symmetric normalization
  k_init_deg<<<g(N, T), T, 0, stream>>>(dinv, (int)N);
  k_accum_deg<<<g(E, T), T, 0, stream>>>(dst, dinv, (int)E);
  k_finalize_dinv<<<g(N, T), T, 0, stream>>>(dinv, (int)N);

  long rowTiles = (N + 15) / 16;

  // Layer 1: H1 = X @ W1 ; AGG1 = b1 + self + scatter ; relu
  k_gemm_wmma<<<g(rowTiles * 8 * 32, T), T, 0, stream>>>(Xh, P1, H1, (int)N, 128);
  k_init_agg<<<g(N * 128, T), T, 0, stream>>>(H1, dinv, b1, AGG1, N, 128);
  k_edge_scatter<<<g(E * 32, T), T, 0, stream>>>(src, dst, dinv, H1, AGG1, (int)E, 128);
  k_relu_f32_to_f16<<<g(N * 128, T), T, 0, stream>>>(AGG1, RELU, N * 128);

  // Layer 2: H2 = relu(AGG1) @ W2 ; AGG2 = b2 + self + scatter
  k_gemm_wmma<<<g(rowTiles * 4 * 32, T), T, 0, stream>>>(RELU, P2, H2, (int)N, 64);
  k_init_agg<<<g(N * 64, T), T, 0, stream>>>(H2, dinv, b2, AGG2, N, 64);
  k_edge_scatter<<<g(E * 32, T), T, 0, stream>>>(src, dst, dinv, H2, AGG2, (int)E, 64);

  // sigmoid + mean over nodes (deterministic two-stage reduction)
  k_sigmoid_partial<<<nbMean, 256, 0, stream>>>(AGG2, part, (int)N);
  k_mean_final<<<1, 64, 0, stream>>>(part, out, nbMean, 1.0f / (float)N);
}
